// SelfKANtentionND_45612552683527
// MI455X (gfx1250) — matrix operand order repaired and checked by reading, not compiled, not run on hardware
//
#include <hip/hip_runtime.h>

// ---------------------------------------------------------------------------
// CDNA5 / gfx1250, wave32. All matmuls via v_wmma_f32_16x16x32_bf16.
// ---------------------------------------------------------------------------

typedef __attribute__((ext_vector_type(16))) __bf16 v16bf;
typedef __attribute__((ext_vector_type(8)))  float  v8f;

#define WMMA_BF16(A, B, C) \
  __builtin_amdgcn_wmma_f32_16x16x32_bf16(false, (A), false, (B), (short)0, (C), false, false)

// A-fragment (16x32 bf16): lane m=lane&15; VGPR-pair j holds K = 2j + 16*half, +1 (matches B)
__device__ __forceinline__ int a_k(int j, int half) {
  return 2 * j + 16 * half;
}
// B-fragment (32x16 bf16): lane n=lane&15; VGPR-pair j holds K = 2j + 16*half, +1
// C/D (16x16 f32): lane n=lane&15; VGPR r holds row m = r + 8*half

// ---------------------------------------------------------------------------
// Weight conversion fp32 -> bf16
// ---------------------------------------------------------------------------
__global__ __launch_bounds__(256) void k_cvt_bf16(const float* __restrict__ s,
                                                  __bf16* __restrict__ d, int n) {
  int i = blockIdx.x * 256 + threadIdx.x;
  if (i < n) d[i] = (__bf16)s[i];
}

// ---------------------------------------------------------------------------
// 1x1 in-projection: att[b,d,n] = sum_c w[d,c] * x[b,c,n] + bias[d]
// x fp32 [4,256,4096]; w bf16 [64,256]; att bf16 [4,64,4096]
// One wave: 16-pixel tile x all 64 output channels (4 accum tiles), K=256.
// ---------------------------------------------------------------------------
__global__ __launch_bounds__(256) void k_conv1x1_in(const float* __restrict__ x,
                                                    const __bf16* __restrict__ w,
                                                    const float* __restrict__ bias,
                                                    __bf16* __restrict__ att) {
  const int lane = threadIdx.x & 31;
  const int wave = threadIdx.x >> 5;
  const int gw   = blockIdx.x * 8 + wave;      // 1024 waves total
  const int b    = gw >> 8;
  const int n0   = (gw & 255) * 16;
  const int ln   = lane & 15, half = lane >> 4;

  v8f acc[4] = {};
  for (int kb = 0; kb < 256; kb += 32) {
    v16bf bx;
#pragma unroll
    for (int j = 0; j < 8; ++j) {
      int k = 2 * j + 16 * half;
      const float* p = x + ((size_t)(b * 256 + kb + k)) * 4096 + n0 + ln;
      bx[2 * j]     = (__bf16)p[0];
      bx[2 * j + 1] = (__bf16)p[4096];
    }
#pragma unroll
    for (int mt = 0; mt < 4; ++mt) {
      v16bf aw;
#pragma unroll
      for (int j = 0; j < 8; ++j) {
        int k = a_k(j, half);
        const __bf16* p = w + (mt * 16 + ln) * 256 + kb + k;
        aw[2 * j] = p[0]; aw[2 * j + 1] = p[1];
      }
      acc[mt] = WMMA_BF16(aw, bx, acc[mt]);
    }
  }
#pragma unroll
  for (int mt = 0; mt < 4; ++mt)
#pragma unroll
    for (int r = 0; r < 8; ++r) {
      int d = mt * 16 + r + 8 * half;
      att[((size_t)(b * 64 + d)) * 4096 + n0 + ln] = (__bf16)(acc[mt][r] + bias[d]);
    }
}

// ---------------------------------------------------------------------------
// 3x3 conv (pad=1) as implicit GEMM: loop over 9 taps x 2 channel-slices of 32.
// src bf16 [4,64,64,64]; w bf16 [64][576] ([o][i*9+t]); dst bf16 [4,64,4096]
// One wave: 16-pixel tile (within one image row) x 64 out channels.
// ---------------------------------------------------------------------------
__global__ __launch_bounds__(256) void k_conv3x3(const __bf16* __restrict__ src,
                                                 const __bf16* __restrict__ w,
                                                 const float* __restrict__ bias,
                                                 __bf16* __restrict__ dst) {
  const int lane = threadIdx.x & 31;
  const int wave = threadIdx.x >> 5;
  const int gw   = blockIdx.x * 8 + wave;
  const int b    = gw >> 8;
  const int tile = gw & 255;
  const int h    = tile >> 2;
  const int w0   = (tile & 3) * 16;
  const int n0   = h * 64 + w0;
  const int ln   = lane & 15, half = lane >> 4;

  v8f acc[4] = {};
  for (int t = 0; t < 9; ++t) {
    const int dy = t / 3 - 1, dx = t % 3 - 1;
    const int hh = h + dy;
    const int col = w0 + ln + dx;
    const bool ok = (hh >= 0) && (hh < 64) && (col >= 0) && (col < 64);
    for (int ib = 0; ib < 64; ib += 32) {
      v16bf bx;
#pragma unroll
      for (int j = 0; j < 8; ++j) {
        int k = 2 * j + 16 * half;
        if (ok) {
          const __bf16* p = src + ((size_t)(b * 64 + ib + k)) * 4096 + hh * 64 + col;
          bx[2 * j] = p[0]; bx[2 * j + 1] = p[4096];
        } else {
          bx[2 * j] = (__bf16)0.0f; bx[2 * j + 1] = (__bf16)0.0f;
        }
      }
#pragma unroll
      for (int mt = 0; mt < 4; ++mt) {
        v16bf aw;
#pragma unroll
        for (int j = 0; j < 8; ++j) {
          int k = a_k(j, half);
          const __bf16* p = w + (mt * 16 + ln) * 576 + (ib + k) * 9 + t;
          aw[2 * j] = p[0]; aw[2 * j + 1] = p[9];
        }
        acc[mt] = WMMA_BF16(aw, bx, acc[mt]);
      }
    }
  }
#pragma unroll
  for (int mt = 0; mt < 4; ++mt)
#pragma unroll
    for (int r = 0; r < 8; ++r) {
      int d = mt * 16 + r + 8 * half;
      dst[((size_t)(b * 64 + d)) * 4096 + n0 + ln] = (__bf16)(acc[mt][r] + bias[d]);
    }
}

// ---------------------------------------------------------------------------
// Flash attention, no 1/sqrt(d) scale (matches reference).
// q,k,v bf16 [4,64,4096] channel-major; o bf16 [4,64,4096].
// One wave: 16 queries; online softmax over 4096 keys in steps of 32.
// S = Q^T K (d=64 -> 2 WMMA k-steps); O += P * V^T (4 channel tiles).
// ---------------------------------------------------------------------------
__global__ __launch_bounds__(256) void k_flash_attn(const __bf16* __restrict__ q,
                                                    const __bf16* __restrict__ kM,
                                                    const __bf16* __restrict__ vM,
                                                    __bf16* __restrict__ o) {
  __shared__ __bf16 plds[8][16][32];   // per-wave P tile (layout shuffle S->A)
  const int lane = threadIdx.x & 31;
  const int wave = threadIdx.x >> 5;
  const int gw   = blockIdx.x * 8 + wave;
  const int b    = gw >> 8;
  const int i0   = (gw & 255) * 16;
  const int ln   = lane & 15, half = lane >> 4;

  // Q^T A-fragments: element (m=query, k=channel)
  v16bf aq[2];
#pragma unroll
  for (int c2 = 0; c2 < 2; ++c2)
#pragma unroll
    for (int j = 0; j < 8; ++j) {
      int c = c2 * 32 + a_k(j, half);
      const __bf16* p = q + ((size_t)(b * 64 + c)) * 4096 + i0 + ln;
      aq[c2][2 * j] = p[0]; aq[c2][2 * j + 1] = p[4096];
    }

  v8f  oacc[4] = {};
  float rmax[8], rsum[8];
#pragma unroll
  for (int r = 0; r < 8; ++r) { rmax[r] = -1e30f; rsum[r] = 0.0f; }

  for (int jb = 0; jb < 4096; jb += 32) {
    // --- S tiles: [16 queries x 32 keys] as two 16x16 f32 tiles
    v8f s[2];
#pragma unroll
    for (int ht = 0; ht < 2; ++ht) {
      const int j0 = jb + ht * 16;
      v8f accS = {};
#pragma unroll
      for (int c2 = 0; c2 < 2; ++c2) {
        v16bf bk;
#pragma unroll
        for (int j = 0; j < 8; ++j) {
          int c = c2 * 32 + 2 * j + 16 * half;
          const __bf16* p = kM + ((size_t)(b * 64 + c)) * 4096 + j0 + ln;
          bk[2 * j] = p[0]; bk[2 * j + 1] = p[4096];
        }
        accS = WMMA_BF16(aq[c2], bk, accS);
      }
      s[ht] = accS;
    }
    // --- online softmax per row (row m = r + 8*half; 16 lanes per half share a row)
    float scale[8];
#pragma unroll
    for (int r = 0; r < 8; ++r) {
      float t = fmaxf(s[0][r], s[1][r]);
#pragma unroll
      for (int d = 8; d >= 1; d >>= 1) t = fmaxf(t, __shfl_xor(t, d, 32));
      float mnew = fmaxf(rmax[r], t);
      scale[r] = __expf(rmax[r] - mnew);
      rmax[r]  = mnew;
      float p0 = __expf(s[0][r] - mnew);
      float p1 = __expf(s[1][r] - mnew);
      float ps = p0 + p1;
#pragma unroll
      for (int d = 8; d >= 1; d >>= 1) ps += __shfl_xor(ps, d, 32);
      rsum[r] = rsum[r] * scale[r] + ps;
      const int m = r + 8 * half;
      plds[wave][m][ln]      = (__bf16)p0;
      plds[wave][m][16 + ln] = (__bf16)p1;
    }
    // rescale running O
#pragma unroll
    for (int ct = 0; ct < 4; ++ct)
#pragma unroll
      for (int r = 0; r < 8; ++r) oacc[ct][r] *= scale[r];

    // wave-local LDS RAW fence (lanes are lockstep within a wave)
    asm volatile("s_wait_dscnt 0x0" ::: "memory");

    // P as A-fragment [16 queries x 32 keys]
    v16bf ap;
#pragma unroll
    for (int j = 0; j < 8; ++j) {
      int kk = a_k(j, half);
      ap[2 * j]     = plds[wave][ln][kk];
      ap[2 * j + 1] = plds[wave][ln][kk + 1];
    }
    // O += P * V^T : B = [32 keys x 16 channels]
#pragma unroll
    for (int ct = 0; ct < 4; ++ct) {
      v16bf bv;
#pragma unroll
      for (int j = 0; j < 8; ++j) {
        int kk = 2 * j + 16 * half;
        const __bf16* p = vM + ((size_t)(b * 64 + ct * 16 + ln)) * 4096 + jb + kk;
        bv[2 * j] = p[0]; bv[2 * j + 1] = p[1];
      }
      oacc[ct] = WMMA_BF16(ap, bv, oacc[ct]);
    }
  }
  // epilogue: normalize and store [b, c, i] channel-major
#pragma unroll
  for (int r = 0; r < 8; ++r) {
    float inv = 1.0f / rsum[r];
#pragma unroll
    for (int ct = 0; ct < 4; ++ct) {
      o[((size_t)(b * 64 + ct * 16 + ln)) * 4096 + i0 + r + 8 * half] =
          (__bf16)(oacc[ct][r] * inv);
    }
  }
}

// ---------------------------------------------------------------------------
// Out-projection (256x64 GEMM) + residual gamma*x + per-pixel channel LayerNorm.
// One block: 16-pixel tile; 8 waves each produce 32 of the 256 channels.
// ---------------------------------------------------------------------------
__global__ __launch_bounds__(256) void k_outproj_ln(const __bf16* __restrict__ ao,
                                                    const __bf16* __restrict__ w,
                                                    const float* __restrict__ b_out,
                                                    const float* __restrict__ x,
                                                    const float* __restrict__ gamma,
                                                    const float* __restrict__ ln_w,
                                                    const float* __restrict__ ln_b,
                                                    float* __restrict__ out) {
  __shared__ float ybuf[256][17];
  const int lane = threadIdx.x & 31;
  const int wave = threadIdx.x >> 5;
  const int b    = blockIdx.x >> 8;
  const int n0   = (blockIdx.x & 255) * 16;
  const int ln   = lane & 15, half = lane >> 4;

  v8f acc[2] = {};
#pragma unroll
  for (int kb = 0; kb < 64; kb += 32) {
    v16bf bx;
#pragma unroll
    for (int j = 0; j < 8; ++j) {
      int k = 2 * j + 16 * half;
      const __bf16* p = ao + ((size_t)(b * 64 + kb + k)) * 4096 + n0 + ln;
      bx[2 * j] = p[0]; bx[2 * j + 1] = p[4096];
    }
#pragma unroll
    for (int u = 0; u < 2; ++u) {
      int mt = wave * 2 + u;
      v16bf aw;
#pragma unroll
      for (int j = 0; j < 8; ++j) {
        int k = a_k(j, half);
        const __bf16* p = w + (mt * 16 + ln) * 64 + kb + k;
        aw[2 * j] = p[0]; aw[2 * j + 1] = p[1];
      }
      acc[u] = WMMA_BF16(aw, bx, acc[u]);
    }
  }
  const float g = gamma[0];
#pragma unroll
  for (int u = 0; u < 2; ++u)
#pragma unroll
    for (int r = 0; r < 8; ++r) {
      int ch = (wave * 2 + u) * 16 + r + 8 * half;
      float y = acc[u][r] + b_out[ch] + g * x[((size_t)(b * 256 + ch)) * 4096 + n0 + ln];
      ybuf[ch][ln] = y;
    }
  __syncthreads();

  // LayerNorm over 256 channels for one pixel, 16 threads per pixel
  const int p   = threadIdx.x >> 4;
  const int g16 = threadIdx.x & 15;
  float s = 0.0f, s2 = 0.0f;
#pragma unroll
  for (int i = 0; i < 16; ++i) {
    float yv = ybuf[g16 + 16 * i][p];
    s += yv; s2 += yv * yv;
  }
#pragma unroll
  for (int d = 8; d >= 1; d >>= 1) { s += __shfl_xor(s, d, 32); s2 += __shfl_xor(s2, d, 32); }
  const float mean = s * (1.0f / 256.0f);
  const float var  = s2 * (1.0f / 256.0f) - mean * mean;
  const float rstd = rsqrtf(var + 1e-5f);
#pragma unroll
  for (int i = 0; i < 16; ++i) {
    int ch = g16 + 16 * i;
    float yv = ybuf[ch][p];
    out[((size_t)(b * 256 + ch)) * 4096 + n0 + p] = (yv - mean) * rstd * ln_w[ch] + ln_b[ch];
  }
}

// ---------------------------------------------------------------------------
// Host launcher
// ---------------------------------------------------------------------------
extern "C" void kernel_launch(void* const* d_in, const int* in_sizes, int n_in,
                              void* d_out, int out_size, void* d_ws, size_t ws_size,
                              hipStream_t stream) {
  const float* x     = (const float*)d_in[0];
  const float* w_in  = (const float*)d_in[1];
  const float* b_in  = (const float*)d_in[2];
  const float* wq    = (const float*)d_in[3];
  const float* bq    = (const float*)d_in[4];
  const float* wk    = (const float*)d_in[5];
  const float* bk    = (const float*)d_in[6];
  const float* wv    = (const float*)d_in[7];
  const float* bv    = (const float*)d_in[8];
  const float* w_out = (const float*)d_in[9];
  const float* b_out = (const float*)d_in[10];
  const float* gamma = (const float*)d_in[11];
  const float* ln_w  = (const float*)d_in[12];
  const float* ln_b  = (const float*)d_in[13];
  float* out = (float*)d_out;

  const size_t MB = 1u << 20;
  char* ws = (char*)d_ws;
  __bf16* att    = (__bf16*)(ws + 0 * MB);   // [4,64,4096]
  __bf16* qb     = (__bf16*)(ws + 2 * MB);
  __bf16* kb     = (__bf16*)(ws + 4 * MB);
  __bf16* vb     = (__bf16*)(ws + 6 * MB);
  __bf16* aob    = (__bf16*)(ws + 8 * MB);
  __bf16* w_in_b = (__bf16*)(ws + 10 * MB);
  __bf16* wq_b   = (__bf16*)(ws + 10 * MB + 32768);
  __bf16* wk_b   = (__bf16*)(ws + 10 * MB + 32768 + 73728);
  __bf16* wv_b   = (__bf16*)(ws + 10 * MB + 32768 + 2 * 73728);
  __bf16* w_out_b= (__bf16*)(ws + 10 * MB + 32768 + 3 * 73728);

  k_cvt_bf16<<<(16384 + 255) / 256, 256, 0, stream>>>(w_in, w_in_b, 16384);
  k_cvt_bf16<<<(36864 + 255) / 256, 256, 0, stream>>>(wq, wq_b, 36864);
  k_cvt_bf16<<<(36864 + 255) / 256, 256, 0, stream>>>(wk, wk_b, 36864);
  k_cvt_bf16<<<(36864 + 255) / 256, 256, 0, stream>>>(wv, wv_b, 36864);
  k_cvt_bf16<<<(16384 + 255) / 256, 256, 0, stream>>>(w_out, w_out_b, 16384);

  k_conv1x1_in<<<128, 256, 0, stream>>>(x, w_in_b, b_in, att);
  k_conv3x3<<<128, 256, 0, stream>>>(att, wq_b, bq, qb);
  k_conv3x3<<<128, 256, 0, stream>>>(att, wk_b, bk, kb);
  k_conv3x3<<<128, 256, 0, stream>>>(att, wv_b, bv, vb);
  k_flash_attn<<<128, 256, 0, stream>>>(qb, kb, vb, aob);
  k_outproj_ln<<<1024, 256, 0, stream>>>(aob, w_out_b, b_out, x, gamma, ln_w, ln_b, out);
}
